// VariationalConfounder_44220983280454
// MI455X (gfx1250) — compile-verified, matrix operand order checked by reading, not run on hardware
//
#include <hip/hip_runtime.h>
#include <hip/hip_bf16.h>

// ---------------------------------------------------------------------------
// Types for CDNA5 WMMA (gfx1250, wave32)
// ---------------------------------------------------------------------------
typedef __bf16 bf16_t;
typedef bf16_t v16bf __attribute__((ext_vector_type(16)));
typedef bf16_t v8bf  __attribute__((ext_vector_type(8)));
typedef float  v8f   __attribute__((ext_vector_type(8)));

#define LEAKY_SLOPE 0.2f

// ---------------------------------------------------------------------------
// Order-preserving float<->uint encoding for atomicMax-based segment max
// ---------------------------------------------------------------------------
__device__ __forceinline__ unsigned f2ord(float f) {
    unsigned u = __float_as_uint(f);
    return (u & 0x80000000u) ? ~u : (u | 0x80000000u);
}
__device__ __forceinline__ float ord2f(unsigned u) {
    return (u & 0x80000000u) ? __uint_as_float(u & 0x7fffffffu)
                             : __uint_as_float(~u);
}

__device__ __forceinline__ v16bf frag16(const bf16_t* p) {
    v8bf lo = *(const v8bf*)(p);
    v8bf hi = *(const v8bf*)(p + 16);
    return __builtin_shufflevector(lo, hi, 0,1,2,3,4,5,6,7,8,9,10,11,12,13,14,15);
}

// ---------------------------------------------------------------------------
// Weight prep: fp32 W[K x N] row-major  ->  bf16 Wt[N x K] row-major
// (transposed so GEMM B-fragment loads are contiguous 16B per lane)
// ---------------------------------------------------------------------------
__global__ void transpose_to_bf16(const float* __restrict__ W, bf16_t* __restrict__ Wt,
                                  int K, int Ncols) {
    int idx = blockIdx.x * blockDim.x + threadIdx.x;
    int total = K * Ncols;
    if (idx >= total) return;
    int k = idx / Ncols, n = idx % Ncols;
    Wt[(size_t)n * K + k] = (bf16_t)W[idx];
}

// ---------------------------------------------------------------------------
// Gather + concat + convert: x = [ts_data | ent_tab[ent_idx] | ts_tab[ts_idx]]
// ---------------------------------------------------------------------------
__global__ void gather_concat_bf16(const float* __restrict__ ts_data,
                                   const int* __restrict__ ent_idx,
                                   const int* __restrict__ ts_idx,
                                   const float* __restrict__ ent_tab,
                                   const float* __restrict__ ts_tab,
                                   bf16_t* __restrict__ xbf, int Nn) {
    size_t idx = (size_t)blockIdx.x * blockDim.x + threadIdx.x;
    size_t total = (size_t)Nn * 192;
    if (idx >= total) return;
    int row = (int)(idx / 192);
    int col = (int)(idx % 192);
    float v;
    if (col < 64)        v = ts_data[(size_t)row * 64 + col];
    else if (col < 128)  v = ent_tab[(size_t)ent_idx[row] * 64 + (col - 64)];
    else                 v = ts_tab[(size_t)ts_idx[row] * 64 + (col - 128)];
    xbf[idx] = (bf16_t)v;
}

// ---------------------------------------------------------------------------
// bf16 GEMM with WMMA, register-blocked 16x64 per wave:
//   C[M x Ncols] = A[M x K] * Bt^T + bias
//   A  : bf16 row-major [M x K]
//   Bt : bf16 row-major [Ncols x K]  (B transposed)
// One wave holds 4 accumulators (4 col-tiles) and reuses one A fragment per
// K-step across 4 v_wmma_f32_16x16x32_bf16 -> 2 A-loads per 4 WMMAs.
// Requires: M % 16 == 0, Ncols % 64 == 0, K % 32 == 0 (true for all calls).
// ---------------------------------------------------------------------------
__global__ void __launch_bounds__(256)
gemm_bf16_wmma(const bf16_t* __restrict__ A, const bf16_t* __restrict__ Bt,
               const float* __restrict__ bias,
               float* __restrict__ C, bf16_t* __restrict__ Cbf,
               int M, int Ncols, int K, int do_relu) {
    int wave = threadIdx.x >> 5;
    int lane = threadIdx.x & 31;
    int colGroups  = Ncols >> 6;                  // groups of 4 col-tiles (64 cols)
    int totalTiles = (M >> 4) * colGroups;
    int tile = blockIdx.x * (blockDim.x >> 5) + wave;  // uniform per wave
    if (tile >= totalTiles) return;                    // whole wave exits: EXEC all-1

    int rowTile  = tile / colGroups;
    int colGroup = tile % colGroups;
    int l15 = lane & 15;
    int kb  = (lane >> 4) << 3;   // 0 for lanes 0-15, 8 for lanes 16-31

    const bf16_t* Ap = A  + (size_t)(rowTile * 16 + l15) * K + kb;
    const bf16_t* Bp = Bt + (size_t)(colGroup * 64 + l15) * K + kb;
    const size_t bstride = (size_t)16 * K;        // next col-tile within group

    v8f acc[4] = {};
    for (int k0 = 0; k0 < K; k0 += 32) {
        v16bf a = frag16(Ap + k0);
#pragma unroll
        for (int j = 0; j < 4; ++j) {
            v16bf b = frag16(Bp + (size_t)j * bstride + k0);
            acc[j] = __builtin_amdgcn_wmma_f32_16x16x32_bf16(
                /*neg_a=*/false, a, /*neg_b=*/false, b,
                /*c_mod=*/(short)0, acc[j], /*reuse_a=*/false, /*reuse_b=*/false);
        }
    }

    // C/D layout: VGPR v of lane -> row (kb + v), col (lane & 15)
    int mbase = rowTile * 16 + kb;
#pragma unroll
    for (int j = 0; j < 4; ++j) {
        int n = colGroup * 64 + j * 16 + l15;
        float bv = bias ? bias[n] : 0.0f;
#pragma unroll
        for (int v = 0; v < 8; ++v) {
            float val = acc[j][v] + bv;
            if (do_relu) val = fmaxf(val, 0.0f);
            size_t o = (size_t)(mbase + v) * Ncols + n;
            if (C)   C[o]   = val;
            if (Cbf) Cbf[o] = (bf16_t)val;
        }
    }
}

// ---------------------------------------------------------------------------
// GATv2 edge pass 1: score[e,h] = sum_d att[h,d] * leakyrelu(xl[src]+xr[dst])
// and running segment max over dst via ordered-uint atomicMax.
// ---------------------------------------------------------------------------
__global__ void edge_score_kernel(const float* __restrict__ xl, const float* __restrict__ xr,
                                  const float* __restrict__ att, const int* __restrict__ ei,
                                  int E, int Ntot, int Hh, int Dd,
                                  float* __restrict__ score, unsigned* __restrict__ smax_u) {
    int idx = blockIdx.x * blockDim.x + threadIdx.x;
    int total = (E + Ntot) * Hh;
    if (idx >= total) return;
    int e = idx / Hh, h = idx % Hh;
    int s, d;
    if (e < E) { s = ei[e]; d = ei[E + e]; } else { s = d = e - E; }  // self loops appended
    int HD = Hh * Dd;
    const float* pl = xl + (size_t)s * HD + h * Dd;
    const float* pr = xr + (size_t)d * HD + h * Dd;
    const float* pa = att + h * Dd;
    float acc = 0.0f;
    for (int dd = 0; dd < Dd; ++dd) {
        float t = pl[dd] + pr[dd];
        t = (t > 0.0f) ? t : LEAKY_SLOPE * t;
        acc += pa[dd] * t;
    }
    score[idx] = acc;
    atomicMax(&smax_u[(size_t)d * Hh + h], f2ord(acc));
}

// ---------------------------------------------------------------------------
// GATv2 edge pass 2: alpha = exp(score - smax[dst]); denom[dst] += alpha
// ---------------------------------------------------------------------------
__global__ void edge_alpha_kernel(const int* __restrict__ ei, int E, int Ntot, int Hh,
                                  float* __restrict__ score,
                                  const unsigned* __restrict__ smax_u,
                                  float* __restrict__ denom) {
    int idx = blockIdx.x * blockDim.x + threadIdx.x;
    int total = (E + Ntot) * Hh;
    if (idx >= total) return;
    int e = idx / Hh, h = idx % Hh;
    int d;
    if (e < E) { d = ei[E + e]; } else { d = e - E; }
    float m = ord2f(smax_u[(size_t)d * Hh + h]);
    float a = __expf(score[idx] - m);
    score[idx] = a;                         // reuse buffer for alpha
    atomicAdd(&denom[(size_t)d * Hh + h], a);
}

// ---------------------------------------------------------------------------
// GATv2 edge pass 2.5: alpha[e,h] /= denom[dst,h]  (hoists the divide out of
// the D-wide scatter loop: one divide per (edge,head) instead of per element)
// ---------------------------------------------------------------------------
__global__ void edge_norm_kernel(const int* __restrict__ ei, int E, int Ntot, int Hh,
                                 float* __restrict__ alpha,
                                 const float* __restrict__ denom) {
    int idx = blockIdx.x * blockDim.x + threadIdx.x;
    int total = (E + Ntot) * Hh;
    if (idx >= total) return;
    int e = idx / Hh, h = idx % Hh;
    int d;
    if (e < E) { d = ei[E + e]; } else { d = e - E; }
    alpha[idx] = alpha[idx] / denom[(size_t)d * Hh + h];
}

// ---------------------------------------------------------------------------
// GATv2 edge pass 3: out[dst,h,:] += alpha_norm[e,h] * xl[src,h,:]
// One block per edge, blockDim == H*D. Pure FMA + global_atomic_add_f32.
// ---------------------------------------------------------------------------
__global__ void edge_aggr_kernel(const float* __restrict__ xl, const float* __restrict__ alpha,
                                 const int* __restrict__ ei,
                                 int E, int Ntot, int Hh, int Dd, float* __restrict__ out) {
    int e = blockIdx.x;
    int t = threadIdx.x;            // 0 .. H*D-1
    int HD = Hh * Dd;
    int h = t / Dd;
    int s, d;
    if (e < E) { s = ei[e]; d = ei[E + e]; } else { s = d = e - E; }
    float w = alpha[(size_t)e * Hh + h];
    atomicAdd(&out[(size_t)d * HD + t], w * xl[(size_t)s * HD + t]);
}

// ---------------------------------------------------------------------------
// Elementwise: v = in + bias[col]; optional relu; write fp32 and/or bf16
// ---------------------------------------------------------------------------
__global__ void bias_act_kernel(const float* __restrict__ in, const float* __restrict__ bias,
                                int Ncols, size_t total, float* __restrict__ outf,
                                bf16_t* __restrict__ outbf, int do_relu) {
    size_t idx = (size_t)blockIdx.x * blockDim.x + threadIdx.x;
    if (idx >= total) return;
    int c = (int)(idx % Ncols);
    float v = in[idx] + bias[c];
    if (do_relu) v = fmaxf(v, 0.0f);
    if (outf)  outf[idx]  = v;
    if (outbf) outbf[idx] = (bf16_t)v;
}

// ---------------------------------------------------------------------------
// VAE reparameterization: q[N,64] -> mean, logvar, z (fp32 to d_out; bf16 z)
// ---------------------------------------------------------------------------
__global__ void vae_z_kernel(const float* __restrict__ q, const float* __restrict__ eps,
                             float* __restrict__ z_out, float* __restrict__ mean_out,
                             float* __restrict__ logvar_out, bf16_t* __restrict__ zbf, int Nn) {
    int idx = blockIdx.x * blockDim.x + threadIdx.x;
    int total = Nn * 32;
    if (idx >= total) return;
    int n = idx / 32, l = idx % 32;
    float mean = q[(size_t)n * 64 + l];
    float logv = q[(size_t)n * 64 + 32 + l];
    float z = mean + __expf(0.5f * logv) * eps[idx];
    z_out[idx] = z;
    mean_out[idx] = mean;
    logvar_out[idx] = logv;
    zbf[idx] = (bf16_t)z;
}

// ---------------------------------------------------------------------------
// Host-side launch
// ---------------------------------------------------------------------------
extern "C" void kernel_launch(void* const* d_in, const int* in_sizes, int n_in,
                              void* d_out, int out_size, void* d_ws, size_t ws_size,
                              hipStream_t stream) {
    const int N = 50000, E = 400000, GIN = 192, Hh = 4, HID = 64;
    const int HD1 = Hh * HID;           // 256
    const int GOUT = 64, LAT = 32;
    const int EN = E + N;               // edges incl. self loops

    const int*   ent_idx = (const int*)d_in[0];
    const float* ts_data = (const float*)d_in[2];
    const int*   ei      = (const int*)d_in[3];
    const int*   ts_idx  = (const int*)d_in[4];
    const float* eps     = (const float*)d_in[5];
    const float* ent_tab = (const float*)d_in[6];
    const float* ts_tab  = (const float*)d_in[7];
    const float* W1l = (const float*)d_in[8];  const float* b1l  = (const float*)d_in[9];
    const float* W1r = (const float*)d_in[10]; const float* b1r  = (const float*)d_in[11];
    const float* a1  = (const float*)d_in[12]; const float* bias1= (const float*)d_in[13];
    const float* W2l = (const float*)d_in[14]; const float* b2l  = (const float*)d_in[15];
    const float* W2r = (const float*)d_in[16]; const float* b2r  = (const float*)d_in[17];
    const float* a2  = (const float*)d_in[18]; const float* bias2= (const float*)d_in[19];
    const float* We1 = (const float*)d_in[20]; const float* be1  = (const float*)d_in[21];
    const float* We2 = (const float*)d_in[22]; const float* be2  = (const float*)d_in[23];
    const float* Wd1 = (const float*)d_in[24]; const float* bd1  = (const float*)d_in[25];
    const float* Wd2 = (const float*)d_in[26]; const float* bd2  = (const float*)d_in[27];
    float* out = (float*)d_out;

    // ---- carve workspace -------------------------------------------------
    char* ws = (char*)d_ws;
    size_t off = 0;
    auto alloc = [&](size_t bytes) -> void* {
        void* p = ws + off;
        off = (off + bytes + 255) & ~(size_t)255;
        return p;
    };
    bf16_t* xbf   = (bf16_t*)alloc((size_t)N * GIN * 2);
    bf16_t* w1lt  = (bf16_t*)alloc((size_t)GIN * HD1 * 2);
    bf16_t* w1rt  = (bf16_t*)alloc((size_t)GIN * HD1 * 2);
    bf16_t* w2lt  = (bf16_t*)alloc((size_t)HD1 * GOUT * 2);
    bf16_t* w2rt  = (bf16_t*)alloc((size_t)HD1 * GOUT * 2);
    bf16_t* we1t  = (bf16_t*)alloc((size_t)GOUT * HID * 2);
    bf16_t* we2t  = (bf16_t*)alloc((size_t)HID * 64 * 2);
    bf16_t* wd1t  = (bf16_t*)alloc((size_t)LAT * HID * 2);
    bf16_t* wd2t  = (bf16_t*)alloc((size_t)HID * 64 * 2);
    float*  xl1   = (float*)alloc((size_t)N * HD1 * 4);
    float*  xr1   = (float*)alloc((size_t)N * HD1 * 4);
    float*  score1= (float*)alloc((size_t)EN * Hh * 4);
    bf16_t* hbf   = (bf16_t*)alloc((size_t)N * HD1 * 2);
    float*  xl2   = (float*)alloc((size_t)N * GOUT * 4);
    float*  xr2   = (float*)alloc((size_t)N * GOUT * 4);
    float*  score2= (float*)alloc((size_t)EN * 4);
    bf16_t* cbf   = (bf16_t*)alloc((size_t)N * GOUT * 2);
    bf16_t* e1bf  = (bf16_t*)alloc((size_t)N * HID * 2);
    float*  q     = (float*)alloc((size_t)N * 64 * 4);
    bf16_t* zbf   = (bf16_t*)alloc((size_t)N * LAT * 2);
    bf16_t* d1bf  = (bf16_t*)alloc((size_t)N * HID * 2);
    // contiguous zero-initialized region:
    size_t zero_begin = off;
    float*    out1   = (float*)alloc((size_t)N * HD1 * 4);
    float*    out2   = (float*)alloc((size_t)N * GOUT * 4);
    float*    denom1 = (float*)alloc((size_t)N * Hh * 4);
    float*    denom2 = (float*)alloc((size_t)N * 4);
    unsigned* smax1  = (unsigned*)alloc((size_t)N * Hh * 4);
    unsigned* smax2  = (unsigned*)alloc((size_t)N * 4);
    size_t zero_bytes = off - zero_begin;
    hipMemsetAsync(ws + zero_begin, 0, zero_bytes, stream);

    const int TPB = 256;
    auto blocks1d = [](size_t n, int tpb) { return (unsigned)((n + tpb - 1) / tpb); };

    // ---- weight prep (fp32 -> bf16 transposed) ---------------------------
    transpose_to_bf16<<<blocks1d((size_t)GIN * HD1, TPB), TPB, 0, stream>>>(W1l, w1lt, GIN, HD1);
    transpose_to_bf16<<<blocks1d((size_t)GIN * HD1, TPB), TPB, 0, stream>>>(W1r, w1rt, GIN, HD1);
    transpose_to_bf16<<<blocks1d((size_t)HD1 * GOUT, TPB), TPB, 0, stream>>>(W2l, w2lt, HD1, GOUT);
    transpose_to_bf16<<<blocks1d((size_t)HD1 * GOUT, TPB), TPB, 0, stream>>>(W2r, w2rt, HD1, GOUT);
    transpose_to_bf16<<<blocks1d((size_t)GOUT * HID, TPB), TPB, 0, stream>>>(We1, we1t, GOUT, HID);
    transpose_to_bf16<<<blocks1d((size_t)HID * 64, TPB), TPB, 0, stream>>>(We2, we2t, HID, 64);
    transpose_to_bf16<<<blocks1d((size_t)LAT * HID, TPB), TPB, 0, stream>>>(Wd1, wd1t, LAT, HID);
    transpose_to_bf16<<<blocks1d((size_t)HID * 64, TPB), TPB, 0, stream>>>(Wd2, wd2t, HID, 64);

    // ---- node features ---------------------------------------------------
    gather_concat_bf16<<<blocks1d((size_t)N * GIN, TPB), TPB, 0, stream>>>(
        ts_data, ent_idx, ts_idx, ent_tab, ts_tab, xbf, N);

    // ---- GATv2 layer 1 ---------------------------------------------------
    {
        int tiles = (N / 16) * (HD1 / 64);
        unsigned gb = (unsigned)((tiles + 7) / 8);
        gemm_bf16_wmma<<<gb, 256, 0, stream>>>(xbf, w1lt, b1l, xl1, nullptr, N, HD1, GIN, 0);
        gemm_bf16_wmma<<<gb, 256, 0, stream>>>(xbf, w1rt, b1r, xr1, nullptr, N, HD1, GIN, 0);
    }
    edge_score_kernel<<<blocks1d((size_t)EN * Hh, TPB), TPB, 0, stream>>>(
        xl1, xr1, a1, ei, E, N, Hh, HID, score1, smax1);
    edge_alpha_kernel<<<blocks1d((size_t)EN * Hh, TPB), TPB, 0, stream>>>(
        ei, E, N, Hh, score1, smax1, denom1);
    edge_norm_kernel<<<blocks1d((size_t)EN * Hh, TPB), TPB, 0, stream>>>(
        ei, E, N, Hh, score1, denom1);
    edge_aggr_kernel<<<(unsigned)EN, HD1, 0, stream>>>(
        xl1, score1, ei, E, N, Hh, HID, out1);
    bias_act_kernel<<<blocks1d((size_t)N * HD1, TPB), TPB, 0, stream>>>(
        out1, bias1, HD1, (size_t)N * HD1, nullptr, hbf, 1);   // relu(h) -> bf16

    // ---- GATv2 layer 2 (H=1, concat=False -> mean over 1 head) -----------
    {
        int tiles = (N / 16) * (GOUT / 64);
        unsigned gb = (unsigned)((tiles + 7) / 8);
        gemm_bf16_wmma<<<gb, 256, 0, stream>>>(hbf, w2lt, b2l, xl2, nullptr, N, GOUT, HD1, 0);
        gemm_bf16_wmma<<<gb, 256, 0, stream>>>(hbf, w2rt, b2r, xr2, nullptr, N, GOUT, HD1, 0);
    }
    edge_score_kernel<<<blocks1d((size_t)EN, TPB), TPB, 0, stream>>>(
        xl2, xr2, a2, ei, E, N, 1, GOUT, score2, smax2);
    edge_alpha_kernel<<<blocks1d((size_t)EN, TPB), TPB, 0, stream>>>(
        ei, E, N, 1, score2, smax2, denom2);
    edge_norm_kernel<<<blocks1d((size_t)EN, TPB), TPB, 0, stream>>>(
        ei, E, N, 1, score2, denom2);
    edge_aggr_kernel<<<(unsigned)EN, GOUT, 0, stream>>>(
        xl2, score2, ei, E, N, 1, GOUT, out2);
    bias_act_kernel<<<blocks1d((size_t)N * GOUT, TPB), TPB, 0, stream>>>(
        out2, bias2, GOUT, (size_t)N * GOUT, nullptr, cbf, 0); // causality -> bf16

    // ---- VAE encoder -----------------------------------------------------
    {
        int tiles = (N / 16) * (HID / 64);
        unsigned gb = (unsigned)((tiles + 7) / 8);
        // e1 = relu(c @ We1 + be1)   (bf16 epilogue only)
        gemm_bf16_wmma<<<gb, 256, 0, stream>>>(cbf, we1t, be1, nullptr, e1bf, N, HID, GOUT, 1);
        // q = e1 @ We2 + be2   [N, 64] fp32
        gemm_bf16_wmma<<<gb, 256, 0, stream>>>(e1bf, we2t, be2, q, nullptr, N, 64, HID, 0);
    }

    // ---- reparameterize: z / mean / logvar straight into d_out -----------
    float* z_out    = out;
    float* mean_out = out + (size_t)N * LAT;
    float* lv_out   = out + (size_t)2 * N * LAT;
    float* xrec_out = out + (size_t)3 * N * LAT;
    vae_z_kernel<<<blocks1d((size_t)N * LAT, TPB), TPB, 0, stream>>>(
        q, eps, z_out, mean_out, lv_out, zbf, N);

    // ---- VAE decoder -----------------------------------------------------
    {
        int tiles = (N / 16) * (HID / 64);
        unsigned gb = (unsigned)((tiles + 7) / 8);
        // d1 = relu(z @ Wd1 + bd1)   (K = 32 -> single WMMA step)
        gemm_bf16_wmma<<<gb, 256, 0, stream>>>(zbf, wd1t, bd1, nullptr, d1bf, N, HID, LAT, 1);
        // x_rec = d1 @ Wd2 + bd2 -> d_out tail
        gemm_bf16_wmma<<<gb, 256, 0, stream>>>(d1bf, wd2t, bd2, xrec_out, nullptr, N, 64, HID, 0);
    }
}
